// GNNAutoEncoder_70978629533940
// MI455X (gfx1250) — matrix operand order repaired and checked by reading, not compiled, not run on hardware
//
#include <hip/hip_runtime.h>
#include <stdint.h>

// ---------------------------------------------------------------------------
// CDNA5 (gfx1250) GraphSAGE autoencoder, bf16 end-to-end:
//  - packed bf16 scatter atomics (global_atomic_pk_add_bf16) for mean-agg
//  - async memory->LDS tile staging (global_load_async_to_lds_b128, ASYNCcnt)
//  - v_wmma_f32_16x16x32_bf16 dual-GEMM (neighbor + root), f32 accumulate
// ---------------------------------------------------------------------------

typedef __bf16 bf16_t;
typedef __attribute__((ext_vector_type(16))) __bf16 v16bf;
typedef __attribute__((ext_vector_type(8)))  float  v8f;

#define TM 64      // macro tile rows
#define TN 128     // macro tile cols
#define TK 32      // k step (WMMA bf16 K)
#define LDA 40     // TK + 8 halves padding (80B row stride keeps 16B alignment)

// ------------------------- CDNA5 inline-asm helpers ------------------------

typedef __attribute__((address_space(3))) const void* lds_cptr_t;

__device__ __forceinline__ uint32_t lds_off_u32(const void* p) {
  return (uint32_t)(uintptr_t)(lds_cptr_t)p;   // generic -> LDS byte offset
}

// async 16B copy: global memory -> LDS (tracked by ASYNCcnt)
__device__ __forceinline__ void async_copy_b128(uint32_t lds_byte_off, const void* gaddr) {
  asm volatile("global_load_async_to_lds_b128 %0, %1, off"
               :: "v"(lds_byte_off), "v"(gaddr) : "memory");
}

__device__ __forceinline__ void wait_async0() {
  asm volatile("s_wait_asynccnt 0x0" ::: "memory");
}

// packed 2 x bf16 atomic add (no return)
__device__ __forceinline__ void atomic_pk_add_bf16(void* addr, uint32_t data) {
  asm volatile("global_atomic_pk_add_bf16 %0, %1, off" :: "v"(addr), "v"(data) : "memory");
}

// ---------------------------- utility kernels ------------------------------

__global__ void fill_zero_f32(float* __restrict__ p, size_t n) {
  size_t i = (size_t)blockIdx.x * blockDim.x + threadIdx.x;
  if (i < n) p[i] = 0.0f;
}

__global__ void degree_kernel(const int* __restrict__ eidx, float* __restrict__ deg, int E) {
  int e = blockIdx.x * blockDim.x + threadIdx.x;
  if (e < E) atomicAdd(&deg[eidx[E + e]], 1.0f);
}

__global__ void invdeg_kernel(const float* __restrict__ deg, float* __restrict__ invd, int n) {
  int i = blockIdx.x * blockDim.x + threadIdx.x;
  if (i < n) invd[i] = 1.0f / fmaxf(deg[i], 1.0f);
}

// fp32 -> bf16, 8 elements per thread (n8 = n/8)
__global__ void cvt_f32_bf16(const float* __restrict__ src, bf16_t* __restrict__ dst, size_t n8) {
  size_t i = (size_t)blockIdx.x * blockDim.x + threadIdx.x;
  if (i >= n8) return;
  const float* s = src + i * 8;
  bf16_t tmp[8];
#pragma unroll
  for (int j = 0; j < 8; ++j) tmp[j] = (bf16_t)s[j];
  *(uint4*)(dst + i * 8) = *(const uint4*)tmp;
}

// W [K,N] fp32 -> WT [N,K] bf16 (coalesced writes); K is a power of two
__global__ void transpose_to_bf16(const float* __restrict__ W, bf16_t* __restrict__ WT,
                                  int shiftK, int N, size_t total) {
  size_t idx = (size_t)blockIdx.x * blockDim.x + threadIdx.x;
  if (idx >= total) return;
  int K = 1 << shiftK;
  int n = (int)(idx >> shiftK);
  int k = (int)(idx & (size_t)(K - 1));
  WT[idx] = (bf16_t)W[(size_t)k * N + n];
}

// scatter-add x[src] -> agg[dst] in bf16; D = 8<<shift feats, 8 feats/thread,
// 4 packed-bf16 atomics per thread
__global__ void scatter_add_bf16(const bf16_t* __restrict__ x, const int* __restrict__ eidx,
                                 bf16_t* __restrict__ agg, int E, int shift) {
  size_t idx = (size_t)blockIdx.x * blockDim.x + threadIdx.x;
  int e = (int)(idx >> shift);
  if (e >= E) return;
  int c = (int)(idx & ((1u << shift) - 1u)) * 8;
  int D = 8 << shift;
  int s = eidx[e];
  int d = eidx[E + e];
  const uint4 v = *(const uint4*)(x + (size_t)s * D + c);
  bf16_t* p = agg + (size_t)d * D + c;
  atomic_pk_add_bf16(p + 0, v.x);
  atomic_pk_add_bf16(p + 2, v.y);
  atomic_pk_add_bf16(p + 4, v.z);
  atomic_pk_add_bf16(p + 6, v.w);
}

// agg[row,:] *= invd[row] in-place (bf16), 8 elements/thread; shift=log2(D/8)
__global__ void scale_rows_bf16(bf16_t* __restrict__ agg, const float* __restrict__ invd,
                                size_t total8, int shift) {
  size_t idx = (size_t)blockIdx.x * blockDim.x + threadIdx.x;
  if (idx >= total8) return;
  float s = invd[idx >> shift];
  uint4 v = *(uint4*)(agg + idx * 8);
  bf16_t* h = (bf16_t*)&v;
#pragma unroll
  for (int j = 0; j < 8; ++j) h[j] = (bf16_t)((float)h[j] * s);
  *(uint4*)(agg + idx * 8) = v;
}

// ------------------------------- WMMA GEMM ---------------------------------
// DUAL=true : OUT = relu( AGG @ WN + X @ WR + BIAS )  (AGG pre-scaled by 1/deg)
// DUAL=false: OUT = relu( X @ WR + BIAS )
// All matrix inputs bf16; weights pre-transposed to [N,K] so both A and B
// tiles are contiguous along K and staged with async b128 copies to LDS.
// 256 threads = 8 wave32s (2x4); each wave owns 32x32 output (2x2 WMMA tiles).

template <bool DUAL, bool OUTBF>
__global__ __launch_bounds__(256)
void sage_gemm_kernel(const bf16_t* __restrict__ X,    // [M,K]
                      const bf16_t* __restrict__ AGG,  // [M,K] (DUAL)
                      const bf16_t* __restrict__ WRT,  // [N,K]
                      const bf16_t* __restrict__ WNT,  // [N,K] (DUAL)
                      const float*  __restrict__ BIAS, // [N]
                      void* __restrict__ OUTP,         // [M,N] bf16 or f32
                      int M, int K, int N) {
  constexpr int NPH = DUAL ? 2 : 1;
  __shared__ bf16_t As[NPH][TM * LDA];   // As[row][k]
  __shared__ bf16_t Bs[NPH][TN * LDA];   // Bs[col][k]

  const int t    = threadIdx.x;
  const int lane = t & 31;
  const int wave = t >> 5;
  const int wm   = wave >> 2;  // 0..1 : 32-row strip
  const int wn_  = wave & 3;   // 0..3 : 32-col strip
  const int mBase = blockIdx.x * TM;
  const int nBase = blockIdx.y * TN;

  // A loader: 8 halves (16B) per thread, row-clamped (rows >= M feed only
  // outputs the epilogue never stores).
  const int ar = t >> 2;           // 0..63
  const int ac = (t & 3) * 8;      // 0,8,16,24
  const int arow = (mBase + ar < M) ? (mBase + ar) : (M - 1);
  // B loader: 16 halves (2x16B) per thread along K of WT.
  const int bcol = t >> 1;         // 0..127
  const int bkc  = (t & 1) * 16;   // 0 / 16

  const bf16_t* aX = X + (size_t)arow * K + ac;
  const bf16_t* aG = DUAL ? (AGG + (size_t)arow * K + ac) : nullptr;
  const bf16_t* bR = WRT + (size_t)(nBase + bcol) * K + bkc;
  const bf16_t* bN = DUAL ? (WNT + (size_t)(nBase + bcol) * K + bkc) : nullptr;

  const uint32_t ldsA0 = lds_off_u32(&As[0][ar * LDA + ac]);
  const uint32_t ldsB0 = lds_off_u32(&Bs[0][bcol * LDA + bkc]);
  uint32_t ldsA1 = 0, ldsB1 = 0;
  if constexpr (DUAL) {
    ldsA1 = lds_off_u32(&As[1][ar * LDA + ac]);
    ldsB1 = lds_off_u32(&Bs[1][bcol * LDA + bkc]);
  }

  v8f acc[2][2] = {};

  const int r15 = lane & 15;
  const int kb  = (lane >> 4) << 3;  // unused: 0 or 8
  const int kb2 = (lane >> 4) << 4;  // B frag K base:     0 or 16

  for (int k = 0; k < K; k += TK) {
    // ---- async stage A/B tiles into LDS (no VGPR round trip) ----
    async_copy_b128(ldsA0, aX + k);
    async_copy_b128(ldsB0, bR + k);
    async_copy_b128(ldsB0 + 16, bR + k + 8);
    if constexpr (DUAL) {
      async_copy_b128(ldsA1, aG + k);
      async_copy_b128(ldsB1, bN + k);
      async_copy_b128(ldsB1 + 16, bN + k + 8);
    }
    wait_async0();
    __syncthreads();

#pragma unroll
    for (int p = 0; p < NPH; ++p) {
      v16bf afr[2], bfr[2];
      // A fragment: lanes 0-15 K=0..15, lanes 16-31 K=16..31 (matches B)
#pragma unroll
      for (int mi = 0; mi < 2; ++mi) {
        const bf16_t* ap = &As[p][(wm * 32 + mi * 16 + r15) * LDA];
#pragma unroll
        for (int i = 0; i < 8; ++i) {
          afr[mi][i]     = ap[kb2 + i];
          afr[mi][8 + i] = ap[kb2 + 8 + i];
        }
      }
      // B fragment (column-major): lanes 0-15 K=0..15, lanes 16-31 K=16..31
#pragma unroll
      for (int ni = 0; ni < 2; ++ni) {
        const bf16_t* bp = &Bs[p][(wn_ * 32 + ni * 16 + r15) * LDA + kb2];
#pragma unroll
        for (int i = 0; i < 16; ++i) bfr[ni][i] = bp[i];
      }
#pragma unroll
      for (int mi = 0; mi < 2; ++mi)
#pragma unroll
        for (int ni = 0; ni < 2; ++ni)
          acc[mi][ni] = __builtin_amdgcn_wmma_f32_16x16x32_bf16(
              false, afr[mi], false, bfr[ni], (short)0, acc[mi][ni], false, false);
    }
    __syncthreads();
  }

  // ---- epilogue: bias + ReLU + store ----
  // C/D layout: vgpr j -> M = j + 8*(lane>=16), N = lane&15. 32-bit index math
  // (M*N < 2^31) and a uniform fast path for full row-blocks (781 of 782).
  const int lrow = (lane >> 4) * 8;
  const bool full = (mBase + TM) <= M;
#pragma unroll
  for (int ni = 0; ni < 2; ++ni) {
    const uint32_t col = (uint32_t)(nBase + wn_ * 32 + ni * 16 + r15);
    const float bv = BIAS[col];
#pragma unroll
    for (int mi = 0; mi < 2; ++mi) {
      const uint32_t rbase = (uint32_t)(mBase + wm * 32 + mi * 16 + lrow);
      uint32_t idx = rbase * (uint32_t)N + col;
      if (full) {
#pragma unroll
        for (int j = 0; j < 8; ++j, idx += (uint32_t)N) {
          const float r = fmaxf(acc[mi][ni][j] + bv, 0.0f);
          if constexpr (OUTBF) ((bf16_t*)OUTP)[idx] = (bf16_t)r;
          else                 ((float*)OUTP)[idx]  = r;
        }
      } else {
#pragma unroll
        for (int j = 0; j < 8; ++j, idx += (uint32_t)N) {
          if (rbase + (uint32_t)j < (uint32_t)M) {
            const float r = fmaxf(acc[mi][ni][j] + bv, 0.0f);
            if constexpr (OUTBF) ((bf16_t*)OUTP)[idx] = (bf16_t)r;
            else                 ((float*)OUTP)[idx]  = r;
          }
        }
      }
    }
  }
}

// ------------------------------- launcher ----------------------------------

extern "C" void kernel_launch(void* const* d_in, const int* in_sizes, int n_in,
                              void* d_out, int out_size, void* d_ws, size_t ws_size,
                              hipStream_t stream) {
  const float* x   = (const float*)d_in[0];
  const int*   ei  = (const int*)d_in[1];   // [2,E]: src row then dst row
  const float* Wn[3] = {(const float*)d_in[2], (const float*)d_in[5], (const float*)d_in[8]};
  const float* Wr[3] = {(const float*)d_in[3], (const float*)d_in[6], (const float*)d_in[9]};
  const float* Bi[3] = {(const float*)d_in[4], (const float*)d_in[7], (const float*)d_in[10]};
  const float* Dw[3] = {(const float*)d_in[11], (const float*)d_in[13], (const float*)d_in[15]};
  const float* Db[3] = {(const float*)d_in[12], (const float*)d_in[14], (const float*)d_in[16]};

  const int Nn = in_sizes[0] / 256;  // 50000 nodes
  const int E  = in_sizes[1] / 2;    // 800000 edges

  // encoder dims: K[i] -> N[i];  decoder dims: dK[i] -> dN[i]
  const int encK[3] = {256, 512, 256}, encN[3] = {512, 256, 128};
  const int decK[3] = {128, 256, 512}, decN[3] = {256, 512, 256};
  const int shK[6]  = {8, 9, 8, 7, 8, 9};  // log2(K) per gemm in order

  // ---- workspace carve (16B-aligned chunks) ----
  float* deg  = (float*)d_ws;
  float* invd = deg + (size_t)((Nn + 63) & ~63);
  bf16_t* xb  = (bf16_t*)(invd + (size_t)((Nn + 63) & ~63));
  bf16_t* bufA = xb + (size_t)Nn * 256;
  bf16_t* bufB = bufA + (size_t)Nn * 512;
  bf16_t* agg  = bufB + (size_t)Nn * 512;
  bf16_t* wp   = agg + (size_t)Nn * 512;
  bf16_t *WrT[3], *WnT[3], *DwT[3];
  for (int i = 0; i < 3; ++i) { WrT[i] = wp; wp += (size_t)encK[i] * encN[i];
                                WnT[i] = wp; wp += (size_t)encK[i] * encN[i]; }
  for (int i = 0; i < 3; ++i) { DwT[i] = wp; wp += (size_t)decK[i] * decN[i]; }

  auto cdiv = [](size_t a, size_t b) { return (int)((a + b - 1) / b); };
  const int MB = cdiv(Nn, TM);

  // ---- one-time-per-call prep: bf16 input, transposed bf16 weights ----
  cvt_f32_bf16<<<cdiv((size_t)Nn * 32, 256), 256, 0, stream>>>(x, xb, (size_t)Nn * 32);
  for (int i = 0; i < 3; ++i) {
    const size_t tot = (size_t)encK[i] * encN[i];
    transpose_to_bf16<<<cdiv(tot, 256), 256, 0, stream>>>(Wr[i], WrT[i], shK[i], encN[i], tot);
    transpose_to_bf16<<<cdiv(tot, 256), 256, 0, stream>>>(Wn[i], WnT[i], shK[i], encN[i], tot);
  }
  for (int i = 0; i < 3; ++i) {
    const size_t tot = (size_t)decK[i] * decN[i];
    transpose_to_bf16<<<cdiv(tot, 256), 256, 0, stream>>>(Dw[i], DwT[i], shK[3 + i], decN[i], tot);
  }

  // ---- degrees ----
  fill_zero_f32<<<cdiv(Nn, 256), 256, 0, stream>>>(deg, (size_t)Nn);
  degree_kernel<<<cdiv(E, 256), 256, 0, stream>>>(ei, deg, E);
  invdeg_kernel<<<cdiv(Nn, 256), 256, 0, stream>>>(deg, invd, Nn);

  // ---- encoder: 3x (scatter-mean + dual WMMA GEMM + ReLU), bf16 out ----
  const bf16_t* hin = xb;
  bf16_t* houts[3] = {bufA, bufB, bufA};
  for (int i = 0; i < 3; ++i) {
    const int D = encK[i];
    const int shift = (D == 512) ? 6 : 5;          // log2(D/8)
    const size_t n8 = (size_t)Nn * (D / 8);
    fill_zero_f32<<<cdiv(n8 * 2, 256), 256, 0, stream>>>((float*)agg, n8 * 2);
    const size_t work = (size_t)E * (D / 8);
    scatter_add_bf16<<<cdiv(work, 256), 256, 0, stream>>>(hin, ei, agg, E, shift);
    scale_rows_bf16<<<cdiv(n8, 256), 256, 0, stream>>>(agg, invd, n8, shift);
    sage_gemm_kernel<true, true><<<dim3(MB, encN[i] / TN), 256, 0, stream>>>(
        hin, agg, WrT[i], WnT[i], Bi[i], houts[i], Nn, D, encN[i]);
    hin = houts[i];
  }

  // ---- decoder: 3x (WMMA GEMM + ReLU); final layer writes fp32 to d_out ----
  sage_gemm_kernel<false, true><<<dim3(MB, decN[0] / TN), 256, 0, stream>>>(
      bufA, nullptr, DwT[0], nullptr, Db[0], bufB, Nn, decK[0], decN[0]);
  sage_gemm_kernel<false, true><<<dim3(MB, decN[1] / TN), 256, 0, stream>>>(
      bufB, nullptr, DwT[1], nullptr, Db[1], bufA, Nn, decK[1], decN[1]);
  sage_gemm_kernel<false, false><<<dim3(MB, decN[2] / TN), 256, 0, stream>>>(
      bufA, nullptr, DwT[2], nullptr, Db[2], d_out, Nn, decK[2], decN[2]);

  (void)n_in; (void)out_size; (void)ws_size;
}
